// MultiHeadAttention_20263655703023
// MI455X (gfx1250) — matrix operand order repaired and checked by reading, not compiled, not run on hardware
//
#include <hip/hip_runtime.h>
#include <hip/hip_bf16.h>

// ---------------------------------------------------------------------------
// Cross-attention (shared sim matrix, softmax over both axes) as TWO flash-
// attention passes with Q/K roles swapped.  bf16 WMMA 16x16x32, f32 accum.
// K/Q tiles staged global->LDS via the Tensor Data Mover, double-buffered so
// the DMA of tile t+1 overlaps WMMA/softmax compute of tile t.
// ---------------------------------------------------------------------------

#define BATCH   16
#define NHEADS  8
#define PADROWS 2
#define HIDDEN  512
#define DH      64
#define SEQ     1024
#define LVALID  (SEQ + PADROWS)   // 1026 real rows (incl. learned pad)
#define LPAD    1088              // 17 tiles of 64
#define NT      17
#define NEG_INF (-1e9f)

typedef __attribute__((ext_vector_type(16))) __bf16 v16bf;
typedef __attribute__((ext_vector_type(8)))  __bf16 v8bf;
typedef __attribute__((ext_vector_type(4)))  __bf16 v4bf;
typedef __attribute__((ext_vector_type(8)))  float  v8f;
typedef __attribute__((ext_vector_type(4)))  unsigned int v4u;
typedef __attribute__((ext_vector_type(8)))  int v8i;
typedef __attribute__((ext_vector_type(4)))  int v4i;

static __device__ __forceinline__ __bf16 f2bf(float f) {
    unsigned u = __float_as_uint(f);
    u += 0x7FFFu + ((u >> 16) & 1u);          // round-to-nearest-even
    unsigned short h = (unsigned short)(u >> 16);
    __bf16 r;
    __builtin_memcpy(&r, &h, sizeof(r));
    return r;
}

// XOR-lane shuffle via ds_swizzle (group-of-32 mode: and=0x1f, or=0, xor=OFF).
template <int OFF>
static __device__ __forceinline__ float swz_xor(float v) {
    return __int_as_float(
        __builtin_amdgcn_ds_swizzle(__float_as_int(v), 0x1f | (OFF << 10)));
}
static __device__ __forceinline__ float red_max16(float v) {
    v = fmaxf(v, swz_xor<1>(v)); v = fmaxf(v, swz_xor<2>(v));
    v = fmaxf(v, swz_xor<4>(v)); v = fmaxf(v, swz_xor<8>(v));
    return v;
}
static __device__ __forceinline__ float red_sum16(float v) {
    v += swz_xor<1>(v); v += swz_xor<2>(v);
    v += swz_xor<4>(v); v += swz_xor<8>(v);
    return v;
}

// B-operand fragment: 16 contiguous bf16 (lane holds N=lane%16, K rows
// kbase2..kbase2+15).  Built from two 16B LDS loads (guaranteed alignment).
static __device__ __forceinline__ v16bf ldB16(const __bf16* p) {
    v8bf a = *(const v8bf*)(p);
    v8bf b = *(const v8bf*)(p + 8);
    return __builtin_shufflevector(a, b, 0,1,2,3,4,5,6,7,8,9,10,11,12,13,14,15);
}
// A-operand fragment: elements 0..15 = K kb..kb+15 (contiguous).
static __device__ __forceinline__ v16bf ldA16(const __bf16* p) {
    v8bf a = *(const v8bf*)(p);
    v8bf b = *(const v8bf*)(p + 8);
    return __builtin_shufflevector(a, b, 0,1,2,3,4,5,6,7,8,9,10,11,12,13,14,15);
}

// ---------------------------------------------------------------------------
// Tensor Data Mover: DMA a 64x64 bf16 row-major tile global -> LDS.
// D# built per CDNA5 ISA 8.3/8.4 (group0: count/lds/global/type=2,
// group1: data_size=2B, tensor 64x64, tile 64x64, stride 64).
// ---------------------------------------------------------------------------
#if __has_builtin(__builtin_amdgcn_tensor_load_to_lds)
#define HAVE_TDM 1
static __device__ __forceinline__ void tdm_load_tile(unsigned ldsOff,
                                                     const __bf16* gptr) {
    unsigned long long ga = (unsigned long long)(uintptr_t)gptr;
    v4u g0 = { 1u,                                   // count=1 (valid, user)
               ldsOff,                               // lds_addr (bytes)
               (unsigned)ga,                         // global_addr[31:0]
               ((unsigned)(ga >> 32) & 0x01FFFFFFu)  // global_addr[56:32]
                   | 0x80000000u };                  // type=2 ("image")
    v8i g1 = { (int)(1u << 16),     // data_size=1 (2 bytes); wg_mask=0
               (int)(64u << 16),    // tensor_dim0 = 64 (bits 79:48 low part)
               (int)(64u << 16),    // tensor_dim0 hi=0 | tensor_dim1 lo=64
               (int)(64u << 16),    // tensor_dim1 hi=0 | tile_dim0=64
               64,                  // tile_dim1=64, tile_dim2=0
               64,                  // tensor_dim0_stride = 64
               0, 0 };              // stride hi / dim1_stride = 0
    v4i z4 = {0, 0, 0, 0};
#if __has_include(<hip/amd_detail/amd_gfx1250_TDM.h>)
    v8i z8 = {0, 0, 0, 0, 0, 0, 0, 0};
    __builtin_amdgcn_tensor_load_to_lds(g0, g1, z4, z4, z8, 0);   // clang-23 arity
#else
    __builtin_amdgcn_tensor_load_to_lds(g0, g1, z4, z4, 0);       // ROCm 7.2 arity
#endif
}
#else
#define HAVE_TDM 0
#endif

// ---------------------------------------------------------------------------
// Pack: fp32 [B,SEQ,512] + pad[2,512] -> bf16 [B,nh,LPAD,64]; plus mask bias.
// ---------------------------------------------------------------------------
__global__ __launch_bounds__(128)
void pack_kernel(const float* __restrict__ src, const float* __restrict__ pad,
                 const long long* __restrict__ mask,
                 __bf16* __restrict__ dst, float* __restrict__ bias) {
    const int b   = blockIdx.x / LPAD;
    const int l   = blockIdx.x % LPAD;
    const int tid = threadIdx.x;
    const int c   = tid * 4;          // column in [0,512)
    const int h   = c >> 6;
    const int d   = c & 63;

    float4 v = make_float4(0.f, 0.f, 0.f, 0.f);
    if (l < PADROWS) {
        v = *(const float4*)(pad + (size_t)l * HIDDEN + c);
    } else if (l < LVALID) {
        v = *(const float4*)(src + ((size_t)b * SEQ + (l - PADROWS)) * HIDDEN + c);
    }
    v4bf o = { f2bf(v.x), f2bf(v.y), f2bf(v.z), f2bf(v.w) };
    *(v4bf*)(dst + (((size_t)b * NHEADS + h) * LPAD + l) * DH + d) = o;

    if (tid == 0) {
        float bv;
        if (l < PADROWS)      bv = 0.f;
        else if (l < LVALID)  bv = mask[(size_t)b * SEQ + (l - PADROWS)] ? 0.f : NEG_INF;
        else                  bv = NEG_INF;
        bias[(size_t)b * LPAD + l] = bv;
    }
}

// ---------------------------------------------------------------------------
// Flash attention: one block = (b, h, 64-row q-tile); 4 waves x 16 q-rows.
// Double-buffered KV tiles: TDM DMA of tile t+1 overlaps compute of tile t.
// ---------------------------------------------------------------------------
__global__ __launch_bounds__(128)
void flash_kernel(const __bf16* __restrict__ Q, const __bf16* __restrict__ K,
                  const float* __restrict__ biasK, float* __restrict__ out) {
    __shared__ __bf16 Qs[64][64];        // q-tile, row-major
    __shared__ __bf16 Ks[2][64][64];     // kv-tile, row-major   (B of Q.K^T)
    __shared__ __bf16 Vt[2][64][80];     // kv-tile, transposed  (B of P.V)
    __shared__ __bf16 Ps[4][16][64];     // per-wave P staging (C->A layout)

    const int qt   = blockIdx.x;      // 0..16
    const int bh   = blockIdx.y;      // 0..127
    const int b    = bh >> 3;
    const int h    = bh & 7;
    const int tid  = threadIdx.x;
    const int wave = tid >> 5;
    const int lane = tid & 31;
    const int ln16 = lane & 15;
    const int hi16 = lane >> 4;       // 0 or 1

    const size_t seqBase = (size_t)bh * LPAD * DH;
    const __bf16* qg = Q + seqBase + (size_t)qt * 64 * DH;

    // per-thread chunk mapping for tile copies (4 x 8 consecutive bf16)
    int crow[4], ccol[4];
#pragma unroll
    for (int i = 0; i < 4; ++i) {
        int chunk = i * 128 + tid;          // 512 chunks x 8 elems
        crow[i] = chunk >> 3;
        ccol[i] = (chunk & 7) * 8;
    }

    // ---- prologue: stage q-tile and kv-tile 0 ----
#if HAVE_TDM
    if (tid < 32) {
        tdm_load_tile((unsigned)(uintptr_t)&Qs[0][0], qg);         // async
        tdm_load_tile((unsigned)(uintptr_t)&Ks[0][0][0], K + seqBase);
    }
#else
#pragma unroll
    for (int i = 0; i < 4; ++i)
        *(v8bf*)(&Qs[crow[i]][ccol[i]]) = *(const v8bf*)(qg + crow[i] * DH + ccol[i]);
#endif
    {
        const __bf16* kg0 = K + seqBase;
#pragma unroll
        for (int i = 0; i < 4; ++i) {
            v8bf vk = *(const v8bf*)(kg0 + crow[i] * DH + ccol[i]);
#if !HAVE_TDM
            *(v8bf*)(&Ks[0][crow[i]][ccol[i]]) = vk;
#endif
#pragma unroll
            for (int e = 0; e < 8; ++e) Vt[0][ccol[i] + e][crow[i]] = vk[e];
        }
    }
#if HAVE_TDM
    if (tid < 32) __builtin_amdgcn_s_wait_tensorcnt(0);            // Q + K0 done
#endif
    __syncthreads();

    // per-wave A fragments for Q (16x32 layout: kb=(lane>=16)*16, contiguous K)
    v16bf aq[2];
    {
        const int kb = hi16 * 16;
#pragma unroll
        for (int k0 = 0; k0 < 2; ++k0)
            aq[k0] = ldA16(&Qs[ln16][k0 * 32 + kb]);
    }

    v8f   accO[4];
    float mrun[8], lrun[8];
#pragma unroll
    for (int j = 0; j < 4; ++j)
#pragma unroll
        for (int r = 0; r < 8; ++r) accO[j][r] = 0.f;
#pragma unroll
    for (int r = 0; r < 8; ++r) { mrun[r] = -3.0e38f; lrun[r] = 0.f; }

    for (int t = 0; t < NT; ++t) {
        const int cur = t & 1;
        const int nxt = cur ^ 1;
        const bool haveNext = (t + 1 < NT);
        const __bf16* kgn = K + seqBase + (size_t)(t + 1) * 64 * DH;

        // ---- kick DMA of next tile; start global reads for next Vt ----
#if HAVE_TDM
        if (haveNext && tid < 32)
            tdm_load_tile((unsigned)(uintptr_t)&Ks[nxt][0][0], kgn);
#endif
        if (t + 2 < NT)   // warm L2 two tiles ahead
            __builtin_prefetch(K + seqBase + (size_t)(t + 2) * 64 * DH
                               + (size_t)tid * 32, 0, 1);
        v8bf vk[4];
        if (haveNext) {
#pragma unroll
            for (int i = 0; i < 4; ++i)
                vk[i] = *(const v8bf*)(kgn + crow[i] * DH + ccol[i]);
        }

        // key-side mask bias for this wave's 4 column tiles
        float bk[4];
#pragma unroll
        for (int j = 0; j < 4; ++j)
            bk[j] = biasK[(size_t)b * LPAD + t * 64 + j * 16 + ln16];

        // ---- S = Q . K^T : preload all 8 B-fragments, then 8 WMMAs ----
        v16bf bS[4][2];
#pragma unroll
        for (int j = 0; j < 4; ++j)
#pragma unroll
            for (int k0 = 0; k0 < 2; ++k0)
                bS[j][k0] = ldB16(&Ks[cur][j * 16 + ln16][k0 * 32 + hi16 * 16]);

        v8f S[4];
#pragma unroll
        for (int j = 0; j < 4; ++j) {
            v8f c = {0.f,0.f,0.f,0.f,0.f,0.f,0.f,0.f};
#pragma unroll
            for (int k0 = 0; k0 < 2; ++k0)
                c = __builtin_amdgcn_wmma_f32_16x16x32_bf16(
                        false, aq[k0], false, bS[j][k0], (short)0, c, false, false);
#pragma unroll
            for (int r = 0; r < 8; ++r) c[r] += bk[j];
            S[j] = c;
        }

        // ---- transpose-scatter next Vt (after S-frag loads: LDS is in-order
        //      per wave, so these stores don't delay the WMMA feed) ----
        if (haveNext) {
#pragma unroll
            for (int i = 0; i < 4; ++i) {
#if !HAVE_TDM
                *(v8bf*)(&Ks[nxt][crow[i]][ccol[i]]) = vk[i];
#endif
#pragma unroll
                for (int e = 0; e < 8; ++e)
                    Vt[nxt][ccol[i] + e][crow[i]] = vk[i][e];
            }
        }

        // ---- online softmax (rows live in VGPR index + lane-half) ----
#pragma unroll
        for (int r = 0; r < 8; ++r) {
            float mx = fmaxf(fmaxf(S[0][r], S[1][r]), fmaxf(S[2][r], S[3][r]));
            mx = red_max16(mx);
            float mnew  = fmaxf(mrun[r], mx);
            float alpha = __expf(mrun[r] - mnew);
            mrun[r] = mnew;
            float rs = 0.f;
#pragma unroll
            for (int j = 0; j < 4; ++j) {
                float p = __expf(S[j][r] - mnew);
                S[j][r] = p;
                rs += p;
            }
            rs = red_sum16(rs);
            lrun[r] = lrun[r] * alpha + rs;
#pragma unroll
            for (int j = 0; j < 4; ++j) accO[j][r] *= alpha;
        }

        // ---- preload P.V B-fragments (independent of P staging) ----
        v16bf bO[4][2];
#pragma unroll
        for (int j = 0; j < 4; ++j)
#pragma unroll
            for (int k0 = 0; k0 < 2; ++k0)
                bO[j][k0] = ldB16(&Vt[cur][j * 16 + ln16][k0 * 32 + hi16 * 16]);

        // ---- C-layout P -> per-wave LDS -> A-layout fragments ----
#pragma unroll
        for (int r = 0; r < 8; ++r) {
            int row = r + hi16 * 8;
#pragma unroll
            for (int j = 0; j < 4; ++j)
                Ps[wave][row][j * 16 + ln16] = f2bf(S[j][r]);
        }
        // wave-local LDS region: DS ops are in-order per wave, no barrier

        // ---- O += P . V ----
#pragma unroll
        for (int k0 = 0; k0 < 2; ++k0) {
            v16bf ap = ldA16(&Ps[wave][ln16][k0 * 32 + hi16 * 16]);
#pragma unroll
            for (int j = 0; j < 4; ++j)
                accO[j] = __builtin_amdgcn_wmma_f32_16x16x32_bf16(
                              false, ap, false, bO[j][k0], (short)0, accO[j],
                              false, false);
        }

        // ---- close the pipeline stage: DMA done, everyone in sync ----
#if HAVE_TDM
        if (haveNext && tid < 32)
            __builtin_amdgcn_s_wait_tensorcnt(0);
#endif
        __syncthreads();
    }

    // ---- normalize + store (drop pad rows / overhang) ----
#pragma unroll
    for (int r = 0; r < 8; ++r) {
        float inv = (lrun[r] > 0.f) ? (1.f / lrun[r]) : 0.f;
        int row = r + hi16 * 8;
        int lq  = qt * 64 + wave * 16 + row;
        if (lq >= PADROWS && lq < LVALID) {
            size_t o = ((size_t)b * SEQ + (lq - PADROWS)) * HIDDEN + h * DH;
#pragma unroll
            for (int j = 0; j < 4; ++j)
                out[o + j * 16 + ln16] = accO[j][r] * inv;
        }
    }
}

// ---------------------------------------------------------------------------
extern "C" void kernel_launch(void* const* d_in, const int* in_sizes, int n_in,
                              void* d_out, int out_size, void* d_ws, size_t ws_size,
                              hipStream_t stream) {
    (void)in_sizes; (void)n_in; (void)out_size; (void)ws_size;

    const float*      X      = (const float*)d_in[0];
    const float*      Y      = (const float*)d_in[1];
    const float*      pad_x  = (const float*)d_in[2];
    const float*      pad_y  = (const float*)d_in[3];
    const long long*  mask_X = (const long long*)d_in[4];
    const long long*  mask_Y = (const long long*)d_in[5];

    char*  ws     = (char*)d_ws;
    size_t qbytes = (size_t)BATCH * NHEADS * LPAD * DH * sizeof(__bf16);
    __bf16* Xq    = (__bf16*)ws;
    __bf16* Yq    = (__bf16*)(ws + qbytes);
    float*  biasX = (float*)(ws + 2 * qbytes);
    float*  biasY = (float*)(ws + 2 * qbytes + (size_t)BATCH * LPAD * sizeof(float));

    float* X_att = (float*)d_out;                                   // [B,N,512]
    float* Y_att = X_att + (size_t)BATCH * SEQ * HIDDEN;            // [B,M,512]

    dim3 pg(BATCH * LPAD);
    pack_kernel<<<pg, 128, 0, stream>>>(X, pad_x, mask_X, Xq, biasX);
    pack_kernel<<<pg, 128, 0, stream>>>(Y, pad_y, mask_Y, Yq, biasY);

    dim3 ag(NT, BATCH * NHEADS);
    // Y_att: row-softmax over keys=Yp  -> Q=Xp, K=V=Yp
    flash_kernel<<<ag, 128, 0, stream>>>(Xq, Yq, biasY, Y_att);
    // X_att: softmax over m == flash with Q=Yp, K=V=Xp
    flash_kernel<<<ag, 128, 0, stream>>>(Yq, Xq, biasX, X_att);
}
